// Correlation_fuse_cat_corre_32392643346645
// MI455X (gfx1250) — matrix operand order repaired and checked by reading, not compiled, not run on hardware
//
#include <hip/hip_runtime.h>
#include <hip/hip_bf16.h>

// Problem dims (fixed by reference)
#define B_   4
#define C_   256
#define H_   64
#define W_   64
#define HW_  4096
#define K_   2304   // C*9 patch features

#define GLOBAL_AS __attribute__((address_space(1)))
#define LDS_AS    __attribute__((address_space(3)))

typedef __attribute__((ext_vector_type(16))) __bf16          v16bf;
typedef __attribute__((ext_vector_type(8)))  float           v8f;
typedef __attribute__((ext_vector_type(8)))  unsigned short  ushort8;
typedef __attribute__((ext_vector_type(4)))  int             v4i;

union Frag {
  ushort8        h[2];
  v16bf          v;
  unsigned short u[16];
};

__device__ __forceinline__ unsigned short f2bf(float f) {
  union { float f; unsigned int u; } x; x.f = f;
  return (unsigned short)(x.u >> 16);
}

// Async global->LDS copy of 16 bytes (ASYNCcnt-tracked on CDNA5)
__device__ __forceinline__ void async_cp16(const unsigned short* g, unsigned short* l) {
#if __has_builtin(__builtin_amdgcn_global_load_async_to_lds_b128)
  __builtin_amdgcn_global_load_async_to_lds_b128(
      (GLOBAL_AS v4i*)g, (LDS_AS v4i*)l, 0, 0);
#else
  unsigned loff = (unsigned)(size_t)(LDS_AS unsigned short*)l;
  asm volatile("global_load_async_to_lds_b128 %0, %1, off"
               :: "v"(loff), "v"(g) : "memory");
#endif
}

__device__ __forceinline__ void wait_async0() {
#if __has_builtin(__builtin_amdgcn_s_wait_asynccnt)
  __builtin_amdgcn_s_wait_asynccnt(0);
#else
  asm volatile("s_wait_asynccnt 0x0" ::: "memory");
#endif
}

// ---------------------------------------------------------------------------
// Click-feature pyramid: maxpool2 + 1x1 conv, three stages
// ---------------------------------------------------------------------------
__global__ __launch_bounds__(256)
void pc1_kernel(const float* __restrict__ click, const float* __restrict__ w1,
                const float* __restrict__ b1, float* __restrict__ out) {
  int idx = blockIdx.x * 256 + threadIdx.x;          // B*64*256*256
  int x = idx & 255, y = (idx >> 8) & 255, o = (idx >> 16) & 63, b = idx >> 22;
  const float* cm = click + (size_t)b * 2 * 512 * 512;
  float acc = b1[o];
  #pragma unroll
  for (int c = 0; c < 2; ++c) {
    const float* p = cm + (size_t)c * 512 * 512 + (size_t)(y * 2) * 512 + x * 2;
    float mv = fmaxf(fmaxf(p[0], p[1]), fmaxf(p[512], p[513]));
    acc += w1[o * 2 + c] * mv;
  }
  out[idx] = acc;
}

__global__ __launch_bounds__(256)
void pc2_kernel(const float* __restrict__ in, const float* __restrict__ w2,
                const float* __restrict__ b2, float* __restrict__ out) {
  int idx = blockIdx.x * 256 + threadIdx.x;          // B*128*128*128
  int x = idx & 127, y = (idx >> 7) & 127, o = (idx >> 14) & 127, b = idx >> 21;
  float acc = b2[o];
  const float* ib = in + (size_t)b * 64 * 256 * 256;
  const float* wr = w2 + o * 64;
  for (int c = 0; c < 64; ++c) {
    const float* p = ib + (size_t)c * 256 * 256 + (size_t)(y * 2) * 256 + x * 2;
    float mv = fmaxf(fmaxf(p[0], p[1]), fmaxf(p[256], p[257]));
    acc += wr[c] * mv;
  }
  out[idx] = acc;
}

__global__ __launch_bounds__(256)
void pc3_kernel(const float* __restrict__ in, const float* __restrict__ w3,
                const float* __restrict__ b3, float* __restrict__ out) {
  int idx = blockIdx.x * 256 + threadIdx.x;          // B*256*64*64
  int n = idx & 4095, o = (idx >> 12) & 255, b = idx >> 20;
  int y = n >> 6, x = n & 63;
  float acc = b3[o];
  const float* ib = in + (size_t)b * 128 * 128 * 128;
  const float* wr = w3 + o * 128;
  for (int c = 0; c < 128; ++c) {
    const float* p = ib + (size_t)c * 128 * 128 + (size_t)(y * 2) * 128 + x * 2;
    float mv = fmaxf(fmaxf(p[0], p[1]), fmaxf(p[128], p[129]));
    acc += wr[c] * mv;
  }
  out[((size_t)b * 256 + o) * 4096 + n] = acc;
}

// cf3 [b][c][n] f32 -> refT [b][n][c] bf16 (V operand, c-contiguous per pixel)
__global__ __launch_bounds__(256)
void reft_kernel(const float* __restrict__ cf, unsigned short* __restrict__ refT) {
  int idx = blockIdx.x * 256 + threadIdx.x;          // B*4096*256
  int c = idx & 255, n = (idx >> 8) & 4095, b = idx >> 20;
  refT[idx] = f2bf(cf[((size_t)b * 256 + c) * 4096 + n]);
}

// ---------------------------------------------------------------------------
// unfold3 + mean-center + L2-normalize -> bf16, [b][pixel][k] k-contiguous.
// One wave32 per pixel; single pass (sum & sumsq), 72 values per lane.
// ---------------------------------------------------------------------------
__global__ __launch_bounds__(128)
void unfold_norm_kernel(const float* __restrict__ fea, unsigned short* __restrict__ out) {
  int wave = threadIdx.x >> 5, lane = threadIdx.x & 31;
  int pix = blockIdx.x * 4 + wave;                   // b*HW + m
  int b = pix >> 12, m = pix & 4095;
  int yy = m >> 6, xx = m & 63;
  const float* fb = fea + (size_t)b * C_ * HW_;
  float vals[72];
  float s = 0.f, sq = 0.f;
  #pragma unroll
  for (int i = 0; i < 72; ++i) {
    int k = lane + i * 32;                           // k = c*9 + p
    int c = k / 9, p = k - c * 9;
    int dy = p / 3 - 1, dx = p % 3 - 1;
    int yu = yy + dy, xu = xx + dx;
    float v = 0.f;
    if (yu >= 0 && yu < H_ && xu >= 0 && xu < W_)
      v = fb[(size_t)c * HW_ + yu * W_ + xu];
    vals[i] = v; s += v; sq += v * v;
  }
  for (int off = 16; off >= 1; off >>= 1) {
    s  += __shfl_xor(s,  off, 32);
    sq += __shfl_xor(sq, off, 32);
  }
  float mean = s * (1.0f / (float)K_);
  float nrm  = sqrtf(fmaxf(sq - (float)K_ * mean * mean, 0.f)) + 2.220446049250313e-16f;
  float inv  = 1.0f / nrm;
  unsigned short* o = out + (size_t)pix * K_;
  #pragma unroll
  for (int i = 0; i < 72; ++i) o[lane + i * 32] = f2bf((vals[i] - mean) * inv);
}

// ---------------------------------------------------------------------------
// Fused flash-attention correlation with async-LDS staging:
//   phi n-tile (32 x K) streamed in 9 k-chunks of 32x256 bf16 via
//   GLOBAL_LOAD_ASYNC_TO_LDS_B128 (double-buffered, shared by 4 waves),
//   S via v_wmma_f32_16x16x32_bf16 (A: Q from global/L2, B: phi from LDS),
//   online softmax, O(16x256) += P x V with V tile also async-staged in LDS.
// LDS rows padded to 264 ushorts (528 B) -> conflict-free b128 B-fragment reads.
// ---------------------------------------------------------------------------
#define KCH   256                 // k-chunk size
#define NKCH  (K_ / KCH)          // 9 chunks
#define LROW  264                 // padded LDS row stride in ushorts

__global__ __launch_bounds__(128)
void attn_kernel(const unsigned short* __restrict__ thetaq,
                 const unsigned short* __restrict__ phik,
                 const unsigned short* __restrict__ refT,
                 float* __restrict__ y) {
  __shared__ alignas(16) unsigned short pbuf[2][32][LROW];  // phi chunk x2
  __shared__ alignas(16) unsigned short vbuf[32][LROW];     // V tile
  __shared__ alignas(16) unsigned short plds[4][16 * 32];   // P bounce / wave

  const int tid  = threadIdx.x;
  const int wave = tid >> 5;
  const int lane = tid & 31;
  const int lrow = lane & 15, lhi = lane >> 4;
  const int mt = blockIdx.x * 4 + wave;              // 1024 m-tiles total
  const int b  = mt >> 8;                            // 256 m-tiles / batch
  const int m0 = (mt & 255) * 16;

  const unsigned short* Q = thetaq + ((size_t)b * HW_ + m0) * K_;
  const unsigned short* P = phik   + (size_t)b * HW_ * K_;
  const unsigned short* V = refT   + (size_t)b * HW_ * C_;

  const float kscale = 100.0f * 1.4426950408889634f;  // (1/TEMP) * log2(e)

  v8f Z = {};
  v8f O[16];
  #pragma unroll
  for (int t = 0; t < 16; ++t) O[t] = Z;
  float mrow[8], lsum[8];
  #pragma unroll
  for (int r = 0; r < 8; ++r) { mrow[r] = -3.0e38f; lsum[r] = 0.f; }

  const unsigned short* qrow = Q + (size_t)lrow * K_ + lhi * 16;

  for (int n0 = 0; n0 < HW_; n0 += 32) {
    // Issue async copies: phi chunk 0 and the V tile (each thread: 8+8 b128)
    #pragma unroll
    for (int i = 0; i < 8; ++i) {
      int j = tid + i * 128;                 // 1024 transfers of 16 B
      int row = j >> 5, seg = j & 31;        // 32 rows x 32 segments
      async_cp16(P + (size_t)(n0 + row) * K_ + seg * 8, &pbuf[0][row][seg * 8]);
    }
    #pragma unroll
    for (int i = 0; i < 8; ++i) {
      int j = tid + i * 128;
      int row = j >> 5, seg = j & 31;
      async_cp16(V + (size_t)(n0 + row) * C_ + seg * 8, &vbuf[row][seg * 8]);
    }

    v8f S0 = {}, S1 = {};
    for (int kc = 0; kc < NKCH; ++kc) {
      wait_async0();
      __syncthreads();                       // staged chunk visible to all waves
      if (kc + 1 < NKCH) {                   // overlap next chunk with compute
        int kbase = (kc + 1) * KCH;
        #pragma unroll
        for (int i = 0; i < 8; ++i) {
          int j = tid + i * 128;
          int row = j >> 5, seg = j & 31;
          async_cp16(P + (size_t)(n0 + row) * K_ + kbase + seg * 8,
                     &pbuf[(kc + 1) & 1][row][seg * 8]);
        }
      }
      const unsigned short* qc = qrow + kc * KCH;
      const unsigned short* pb = &pbuf[kc & 1][0][0];
      const unsigned short* l0 = pb + (size_t)lrow * LROW + lhi * 16;
      const unsigned short* l1 = pb + (size_t)(lrow + 16) * LROW + lhi * 16;
      #pragma unroll 2
      for (int k0 = 0; k0 < KCH; k0 += 32) {
        Frag a, b0, b1;
        a.h[0]  = *(const ushort8*)(qc + k0);
        a.h[1]  = *(const ushort8*)(qc + k0 + 8);
        b0.h[0] = *(const ushort8*)(l0 + k0);
        b0.h[1] = *(const ushort8*)(l0 + k0 + 8);
        b1.h[0] = *(const ushort8*)(l1 + k0);
        b1.h[1] = *(const ushort8*)(l1 + k0 + 8);
        S0 = __builtin_amdgcn_wmma_f32_16x16x32_bf16(false, a.v, false, b0.v,
                                                     (short)0, S0, false, false);
        S1 = __builtin_amdgcn_wmma_f32_16x16x32_bf16(false, a.v, false, b1.v,
                                                     (short)0, S1, false, false);
      }
      __syncthreads();                       // buffer free before re-stage
    }

    // Online softmax over the 32 columns of this tile.
    // C/D layout: VGPR r, lanes 0-15 -> (row r, col lane); lanes 16-31 -> (row r+8, col lane-16).
    #pragma unroll
    for (int r = 0; r < 8; ++r) {
      float tmax = fmaxf(S0[r], S1[r]);
      for (int off = 8; off >= 1; off >>= 1)
        tmax = fmaxf(tmax, __shfl_xor(tmax, off, 16));
      float mnew  = fmaxf(mrow[r], tmax);
      float alpha = exp2f((mrow[r] - mnew) * kscale);
      float e0    = exp2f((S0[r] - mnew) * kscale);
      float e1    = exp2f((S1[r] - mnew) * kscale);
      float rs = e0 + e1;
      for (int off = 8; off >= 1; off >>= 1)
        rs += __shfl_xor(rs, off, 16);
      lsum[r] = lsum[r] * alpha + rs;
      mrow[r] = mnew;
      #pragma unroll
      for (int t = 0; t < 16; ++t) O[t][r] *= alpha;
      int row = r + 8 * lhi;
      plds[wave][row * 32 + lrow]      = f2bf(e0);
      plds[wave][row * 32 + lrow + 16] = f2bf(e1);
    }
    asm volatile("s_wait_dscnt 0x0" ::: "memory");

    // P tile back from LDS as an A-fragment (row-per-lane, k-contiguous)
    Frag pf;
    pf.h[0] = *(const ushort8*)&plds[wave][lrow * 32 + lhi * 16];
    pf.h[1] = *(const ushort8*)&plds[wave][lrow * 32 + lhi * 16 + 8];

    // O += P(16x32) x V(32x256): 16 column-tiles of 16 channels, V from LDS
    const unsigned short* vrow = &vbuf[lhi * 16][lrow];
    #pragma unroll
    for (int t = 0; t < 16; ++t) {
      Frag vf;
      #pragma unroll
      for (int s = 0; s < 16; ++s) vf.u[s] = vrow[s * LROW + t * 16];
      O[t] = __builtin_amdgcn_wmma_f32_16x16x32_bf16(false, pf.v, false, vf.v,
                                                     (short)0, O[t], false, false);
    }
    __syncthreads();                         // V tile free before next re-stage
  }

  // Finalize: divide by row sums, scatter to y[b][c][m] (f32)
  #pragma unroll
  for (int r = 0; r < 8; ++r) {
    float inv = 1.0f / lsum[r];
    int m = m0 + r + 8 * lhi;
    #pragma unroll
    for (int t = 0; t < 16; ++t) {
      int c = t * 16 + lrow;
      y[((size_t)b * C_ + c) * HW_ + m] = O[t][r] * inv;
    }
  }
}

// ---------------------------------------------------------------------------
// Final 1x1 conv: out = wc . concat(block_fea, y) + bc   (512 -> 256)
// ---------------------------------------------------------------------------
__global__ __launch_bounds__(256)
void final_conv_kernel(const float* __restrict__ fea, const float* __restrict__ y,
                       const float* __restrict__ wc, const float* __restrict__ bc,
                       float* __restrict__ out) {
  int blk = blockIdx.x;
  int mb = blk & 15, o = (blk >> 4) & 255, b = blk >> 12;
  int m = mb * 256 + threadIdx.x;
  const float* wrow = wc + o * 512;
  const float* f  = fea + (size_t)b * C_ * HW_ + m;
  const float* yb = y   + (size_t)b * C_ * HW_ + m;
  float acc = bc[o];
  #pragma unroll 4
  for (int c = 0; c < 256; ++c) {
    acc += wrow[c]       * f[(size_t)c * HW_];
    acc += wrow[256 + c] * yb[(size_t)c * HW_];
  }
  out[((size_t)b * C_ + o) * HW_ + m] = acc;
}

// ---------------------------------------------------------------------------
extern "C" void kernel_launch(void* const* d_in, const int* in_sizes, int n_in,
                              void* d_out, int out_size, void* d_ws, size_t ws_size,
                              hipStream_t stream) {
  const float* block_fea = (const float*)d_in[0];
  const float* spp_input = (const float*)d_in[1];
  const float* click_map = (const float*)d_in[2];
  const float* w1 = (const float*)d_in[3];
  const float* b1 = (const float*)d_in[4];
  const float* w2 = (const float*)d_in[5];
  const float* b2 = (const float*)d_in[6];
  const float* w3 = (const float*)d_in[7];
  const float* b3 = (const float*)d_in[8];
  const float* wc = (const float*)d_in[9];
  const float* bc = (const float*)d_in[10];
  float* out = (float*)d_out;

  char* w = (char*)d_ws;
  float*          cf1    = (float*)(w);                         // 4*64*256*256 f32  (64 MiB)
  float*          cf2    = (float*)(w + 67108864);              // 4*128*128*128 f32 (32 MiB)
  float*          cf3    = (float*)(w + 100663296);             // 4*256*4096 f32    (16 MiB)
  unsigned short* thetaq = (unsigned short*)(w + 117440512);    // 4*4096*2304 bf16  (72 MiB)
  unsigned short* phik   = (unsigned short*)(w + 192937984);    // 4*4096*2304 bf16  (72 MiB)
  unsigned short* refT   = (unsigned short*)(w + 268435456);    // 4*4096*256 bf16   ( 8 MiB)
  float*          ybuf   = (float*)(w + 276824064);             // 4*256*4096 f32    (16 MiB)

  pc1_kernel<<<65536, 256, 0, stream>>>(click_map, w1, b1, cf1);
  pc2_kernel<<<32768, 256, 0, stream>>>(cf1, w2, b2, cf2);
  pc3_kernel<<<16384, 256, 0, stream>>>(cf2, w3, b3, cf3);
  reft_kernel<<<16384, 256, 0, stream>>>(cf3, refT);

  unfold_norm_kernel<<<4096, 128, 0, stream>>>(block_fea, thetaq);
  unfold_norm_kernel<<<4096, 128, 0, stream>>>(spp_input, phik);

  attn_kernel<<<256, 128, 0, stream>>>(thetaq, phik, refT, ybuf);

  final_conv_kernel<<<16384, 256, 0, stream>>>(block_fea, ybuf, wc, bc, out);
}
